// SelectiveSSMBlock_34583076667829
// MI455X (gfx1250) — matrix operand order repaired and checked by reading, not compiled, not run on hardware
//
#include <hip/hip_runtime.h>
#include <hip/hip_bf16.h>

// ---------------------------------------------------------------------------
// CDNA5 (gfx1250) fused SelectiveSSMBlock.
//  - bf16 WMMA (v_wmma_f32_16x16x32_bf16) GEMMs, f32 accumulation
//  - weights pre-transposed to [N][K] bf16 once per launch
//  - 128x256x64 block tile, 64x64 wave tile (acc 4x4), double-buffered
//    dynamic LDS (110.6 KB of the 320 KB WGP pool), register prefetch,
//    one barrier per K-tile
//  - fused epilogues; dead `delta` GEMM eliminated
// ---------------------------------------------------------------------------

typedef __bf16 bf16;
typedef __attribute__((ext_vector_type(16))) __bf16 v16bf;
typedef __attribute__((ext_vector_type(8)))  __bf16 v8bf;
typedef __attribute__((ext_vector_type(8)))  float  v8f;
typedef __attribute__((ext_vector_type(4)))  float  v4f;

#define B_DIM   16384
#define DM      1024
#define DS      256

// GEMM tiling
#define BLK_M   128
#define BLK_N   256
#define BLK_K   64
#define LS      72                       // padded LDS row stride (144B = 36 banks)
#define A_TILE  (BLK_M * LS)             // elems per A buffer
#define B_TILE  (BLK_N * LS)             // elems per B buffer
#define GEMM_LDS_BYTES ((2 * A_TILE + 2 * B_TILE) * 2)

union Frag16 { v16bf v; v8bf h[2]; };

__device__ inline v8f zero_v8f() {
  v8f z;
#pragma unroll
  for (int i = 0; i < 8; ++i) z[i] = 0.0f;
  return z;
}

// ---------------- f32 -> bf16 (no transpose; for A which is already [n][k]) -
__global__ void cvt_f32_bf16(const float* __restrict__ s, bf16* __restrict__ d, int n) {
  int i = blockIdx.x * blockDim.x + threadIdx.x;
  if (i < n) d[i] = (bf16)s[i];
}

// ---------------- f32 [K][N] -> bf16 [N][K] transpose-convert ---------------
__global__ __launch_bounds__(256) void transpose_cvt(
    const float* __restrict__ src, bf16* __restrict__ dst, int K, int N) {
  __shared__ float tile[32][33];
  const int kb = blockIdx.y * 32, nb = blockIdx.x * 32;
  const int tx = threadIdx.x & 31, ty = threadIdx.x >> 5;   // 32 x 8
#pragma unroll
  for (int j = 0; j < 32; j += 8)
    tile[ty + j][tx] = src[(size_t)(kb + ty + j) * N + nb + tx];
  __syncthreads();
#pragma unroll
  for (int j = 0; j < 32; j += 8)
    dst[(size_t)(nb + ty + j) * K + kb + tx] = (bf16)tile[tx][ty + j];
}

// ---------------- LayerNorm -> bf16 -----------------------------------------
__global__ __launch_bounds__(256) void layernorm_kernel(
    const float* __restrict__ x, const float* __restrict__ gamma,
    const float* __restrict__ beta, bf16* __restrict__ out) {
  const int row = blockIdx.x;
  const int t   = threadIdx.x;                 // 256 threads, 4 floats each
  const v4f v = reinterpret_cast<const v4f*>(x + (size_t)row * DM)[t];
  float s  = v[0] + v[1] + v[2] + v[3];
  float s2 = v[0]*v[0] + v[1]*v[1] + v[2]*v[2] + v[3]*v[3];
#pragma unroll
  for (int o = 16; o > 0; o >>= 1) {           // wave32 butterfly reduce
    s  += __shfl_xor(s,  o, 32);
    s2 += __shfl_xor(s2, o, 32);
  }
  __shared__ float red[16];
  const int wid = t >> 5, lane = t & 31;
  if (lane == 0) { red[wid] = s; red[8 + wid] = s2; }
  __syncthreads();
  if (t == 0) {
    float a = 0.f, b = 0.f;
    for (int i = 0; i < 8; ++i) { a += red[i]; b += red[8 + i]; }
    red[0] = a; red[8] = b;
  }
  __syncthreads();
  const float mu   = red[0] * (1.0f / DM);
  const float var  = red[8] * (1.0f / DM) - mu * mu;
  const float rstd = rsqrtf(var + 1e-5f);
  const int base = t * 4;
#pragma unroll
  for (int j = 0; j < 4; ++j) {
    float xn = (v[j] - mu) * rstd * gamma[base + j] + beta[base + j];
    out[(size_t)row * DM + base + j] = (bf16)xn;
  }
}

// ---------------- WMMA GEMM with fused epilogues ----------------------------
// C = A(MxK,bf16,[m][k]) * B(KxN,bf16 supplied transposed as Bt[n][k])
// Block tile 128x256x64; 8 waves 2(M)x4(N); each wave 64x64 = 4x4 WMMA tiles.
// Double-buffered dynamic LDS + register prefetch; one barrier per K-tile.
enum { EP_XZ = 0, EP_BX, EP_TANH, EP_Y, EP_OUT };

template <int EP>
__global__ __launch_bounds__(256) void gemm_wmma(
    const bf16* __restrict__ Ag, const bf16* __restrict__ Bt,
    int M, int N, int K,
    const float* __restrict__ bias,
    const float* __restrict__ auxF,   // EP_Y: D vector ; EP_OUT: residual x
    const bf16*  __restrict__ auxB0,  // EP_Y: x_val (bf16)
    const bf16*  __restrict__ auxB1,  // EP_Y: gate  (bf16)
    float* __restrict__ outF,
    bf16*  __restrict__ outB0,
    bf16*  __restrict__ outB1) {
  extern __shared__ bf16 smem[];      // [As0 | As1 | Bs0 | Bs1]

  const int t    = threadIdx.x;
  const int lane = t & 31, wid = t >> 5;
  const int wm = (wid & 1) * 64;       // wave M offset in tile
  const int wn = (wid >> 1) * 64;      // wave N offset in tile
  const int lr = lane & 15, kh = lane >> 4;

  const int n0 = blockIdx.x * BLK_N;
  const int m0 = blockIdx.y * BLK_M;

  v8f acc[4][4];
#pragma unroll
  for (int mi = 0; mi < 4; ++mi)
#pragma unroll
    for (int ni = 0; ni < 4; ++ni) acc[mi][ni] = zero_v8f();

  v8bf ra[4], rb[8];
  // A: 128x64 = 8192 elems -> 4 x b128/thread ; B: 256x64 -> 8 x b128/thread
  auto fill = [&](int kt) {
    const int kbase = kt * BLK_K;
#pragma unroll
    for (int i = 0; i < 4; ++i) {
      const int c = t + i * 256;
      const int row = c >> 3, kc = (c & 7) * 8;
      ra[i] = *reinterpret_cast<const v8bf*>(Ag + (size_t)(m0 + row) * K + kbase + kc);
    }
#pragma unroll
    for (int i = 0; i < 8; ++i) {
      const int c = t + i * 256;
      const int row = c >> 3, kc = (c & 7) * 8;
      rb[i] = *reinterpret_cast<const v8bf*>(Bt + (size_t)(n0 + row) * K + kbase + kc);
    }
  };
  auto store_lds = [&](int buf) {
    bf16* Ab = smem + buf * A_TILE;
    bf16* Bb = smem + 2 * A_TILE + buf * B_TILE;
#pragma unroll
    for (int i = 0; i < 4; ++i) {
      const int c = t + i * 256;
      const int row = c >> 3, kc = (c & 7) * 8;
      *reinterpret_cast<v8bf*>(&Ab[row * LS + kc]) = ra[i];
    }
#pragma unroll
    for (int i = 0; i < 8; ++i) {
      const int c = t + i * 256;
      const int row = c >> 3, kc = (c & 7) * 8;
      *reinterpret_cast<v8bf*>(&Bb[row * LS + kc]) = rb[i];
    }
  };
  auto compute = [&](int buf) {
    const bf16* Ab = smem + buf * A_TILE;
    const bf16* Bb = smem + 2 * A_TILE + buf * B_TILE;
#pragma unroll
    for (int ks = 0; ks < 2; ++ks) {          // two K=32 steps per tile
      Frag16 bfrag[4];
#pragma unroll
      for (int ni = 0; ni < 4; ++ni) {
        const int col = wn + ni * 16 + lr;
        bfrag[ni].h[0] = *reinterpret_cast<const v8bf*>(&Bb[col * LS + ks * 32 + kh * 8]);
        bfrag[ni].h[1] = *reinterpret_cast<const v8bf*>(&Bb[col * LS + ks * 32 + 16 + kh * 8]);
      }
#pragma unroll
      for (int mi = 0; mi < 4; ++mi) {
        const int row = wm + mi * 16 + lr;
        Frag16 afrag;
        afrag.h[0] = *reinterpret_cast<const v8bf*>(&Ab[row * LS + ks * 32 + kh * 8]);
        afrag.h[1] = *reinterpret_cast<const v8bf*>(&Ab[row * LS + ks * 32 + 16 + kh * 8]);
#pragma unroll
        for (int ni = 0; ni < 4; ++ni)
          acc[mi][ni] = __builtin_amdgcn_wmma_f32_16x16x32_bf16(
              false, afrag.v, false, bfrag[ni].v, (short)0, acc[mi][ni], false, false);
      }
    }
  };

  const int nk = K / BLK_K;
  fill(0);
  store_lds(0);
  __syncthreads();
  for (int i = 0; i < nk; ++i) {
    const int cur = i & 1;
    if (i + 1 < nk) fill(i + 1);              // prefetch next tile (in flight)
    compute(cur);
    if (i + 1 < nk) store_lds(cur ^ 1);
    __syncthreads();
  }

  // ---- epilogue: elem r -> M = mi*16 + kh*8 + r ; N = ni*16 + lr
  auto forEachAcc = [&](auto f) {
#pragma unroll
    for (int mi = 0; mi < 4; ++mi)
#pragma unroll
      for (int ni = 0; ni < 4; ++ni)
#pragma unroll
        for (int r = 0; r < 8; ++r) {
          const int m = m0 + wm + mi * 16 + kh * 8 + r;
          const int n = n0 + wn + ni * 16 + lr;
          f(m, n, acc[mi][ni][r]);
        }
  };

  if constexpr (EP == EP_XZ) {
    if (n0 < DM) {                       // whole block in gate half: SiLU
      forEachAcc([&](int m, int n, float v) {
        v += bias[n];
        outB0[(size_t)m * DM + n] = (bf16)(v / (1.0f + __expf(-v)));
      });
    } else {                             // whole block in value half
      forEachAcc([&](int m, int n, float v) {
        outB1[(size_t)m * DM + (n - DM)] = (bf16)(v + bias[n]);
      });
    }
  } else if constexpr (EP == EP_BX) {
    forEachAcc([&](int m, int n, float v) {
      outB0[(size_t)m * N + n] = (bf16)(v + bias[n]);
    });
  } else if constexpr (EP == EP_TANH) {
    forEachAcc([&](int m, int n, float v) {
      outB0[(size_t)m * N + n] = (bf16)tanhf(v);
    });
  } else if constexpr (EP == EP_Y) {
    forEachAcc([&](int m, int n, float v) {
      const float xv = (float)auxB0[(size_t)m * N + n];
      const float g  = (float)auxB1[(size_t)m * N + n];
      outB0[(size_t)m * N + n] = (bf16)((v + bias[n] + auxF[n] * xv) * g);
    });
  } else {  // EP_OUT
    forEachAcc([&](int m, int n, float v) {
      outF[(size_t)m * N + n] = v + bias[n] + auxF[(size_t)m * N + n];
    });
  }
}

// ---------------------------------------------------------------------------
extern "C" void kernel_launch(void* const* d_in, const int* in_sizes, int n_in,
                              void* d_out, int out_size, void* d_ws, size_t ws_size,
                              hipStream_t stream) {
  (void)in_sizes; (void)n_in; (void)out_size; (void)ws_size;
  const float* x     = (const float*)d_in[0];
  const float* W_in  = (const float*)d_in[1];
  const float* b_in  = (const float*)d_in[2];
  const float* A     = (const float*)d_in[3];
  const float* W_B   = (const float*)d_in[4];
  const float* b_B   = (const float*)d_in[5];
  const float* W_C   = (const float*)d_in[6];
  const float* b_C   = (const float*)d_in[7];
  const float* Dv    = (const float*)d_in[8];
  // d_in[9] = W_delta, d_in[10] = b_delta: dead in reference output -> skipped
  const float* W_out = (const float*)d_in[11];
  const float* b_out = (const float*)d_in[12];
  const float* gamma = (const float*)d_in[13];
  const float* beta  = (const float*)d_in[14];
  float* out = (float*)d_out;

  char* ws = (char*)d_ws;
  size_t off = 0;
  auto alloc = [&](size_t bytes) -> char* {
    char* p = ws + off;
    off += (bytes + 255) & ~(size_t)255;
    return p;
  };
  bf16* xn    = (bf16*)alloc((size_t)B_DIM * DM * 2);
  bf16* xval  = (bf16*)alloc((size_t)B_DIM * DM * 2);
  bf16* gate  = (bf16*)alloc((size_t)B_DIM * DM * 2);
  bf16* Bx    = (bf16*)alloc((size_t)B_DIM * DS * 2);
  bf16* h     = (bf16*)alloc((size_t)B_DIM * DS * 2);
  bf16* y2    = xn;   // xn dead after GEMM1 -> reuse for gated y
  // weights, all stored transposed [N][K] except A (used as [n][k] already)
  bf16* W_in_t  = (bf16*)alloc((size_t)DM * 2 * DM * 2);
  bf16* W_B_t   = (bf16*)alloc((size_t)DM * DS * 2);
  bf16* A_b     = (bf16*)alloc((size_t)DS * DS * 2);
  bf16* W_C_t   = (bf16*)alloc((size_t)DS * DM * 2);
  bf16* W_out_t = (bf16*)alloc((size_t)DM * DM * 2);

  // [K][N] f32 -> [N][K] bf16
  auto tcvt = [&](const float* s, bf16* d, int K, int N) {
    transpose_cvt<<<dim3(N / 32, K / 32), 256, 0, stream>>>(s, d, K, N);
  };
  tcvt(W_in,  W_in_t,  DM, 2 * DM);
  tcvt(W_B,   W_B_t,   DM, DS);
  tcvt(W_C,   W_C_t,   DS, DM);
  tcvt(W_out, W_out_t, DM, DM);
  // GEMM3 needs Bt[n][k] = (A^T)[k][n]^T = A row-major: plain convert
  cvt_f32_bf16<<<(DS * DS + 255) / 256, 256, 0, stream>>>(A, A_b, DS * DS);

  layernorm_kernel<<<B_DIM, 256, 0, stream>>>(x, gamma, beta, xn);

  const dim3 blk(256);
  const size_t lds = GEMM_LDS_BYTES;   // 110,592 B (dynamic; 320KB WGP pool)
  // 1) xz = xn @ W_in + b_in ; SiLU gate half, value half -> bf16
  gemm_wmma<EP_XZ><<<dim3((2 * DM) / BLK_N, B_DIM / BLK_M), blk, lds, stream>>>(
      xn, W_in_t, B_DIM, 2 * DM, DM, b_in, nullptr, nullptr, nullptr,
      nullptr, gate, xval);
  // 2) Bx = xval @ W_B + b_B
  gemm_wmma<EP_BX><<<dim3(DS / BLK_N, B_DIM / BLK_M), blk, lds, stream>>>(
      xval, W_B_t, B_DIM, DS, DM, b_B, nullptr, nullptr, nullptr,
      nullptr, Bx, nullptr);
  // 3) h = tanh(Bx @ A^T)
  gemm_wmma<EP_TANH><<<dim3(DS / BLK_N, B_DIM / BLK_M), blk, lds, stream>>>(
      Bx, A_b, B_DIM, DS, DS, nullptr, nullptr, nullptr, nullptr,
      nullptr, h, nullptr);
  // 4) y2 = (h @ W_C + b_C + D * xval) * gate
  gemm_wmma<EP_Y><<<dim3(DM / BLK_N, B_DIM / BLK_M), blk, lds, stream>>>(
      h, W_C_t, B_DIM, DM, DS, b_C, Dv, xval, gate,
      nullptr, y2, nullptr);
  // 5) out = y2 @ W_out + b_out + residual(x)  (f32)
  gemm_wmma<EP_OUT><<<dim3(DM / BLK_N, B_DIM / BLK_M), blk, lds, stream>>>(
      y2, W_out_t, B_DIM, DM, DM, b_out, x, nullptr, nullptr,
      out, nullptr, nullptr);
}
